// MultiHeadSelfAttention_up_44272522887184
// MI455X (gfx1250) — matrix operand order repaired and checked by reading, not compiled, not run on hardware
//
#include <hip/hip_runtime.h>
#include <hip/hip_bf16.h>

typedef __attribute__((ext_vector_type(16))) __bf16 v16bf;
typedef __attribute__((ext_vector_type(8)))  __bf16 v8bf;
typedef __attribute__((ext_vector_type(8)))  float  v8f;
typedef __attribute__((ext_vector_type(4)))  float  v4f;

// B=4, S=1024, D=1024, H=16, dk=64
#define SEQ   1024
#define DDIM  1024
#define NHEAD 16
#define DK    64

__device__ __forceinline__ v16bf ld16bf(const __bf16* p) {
  v8bf lo = *(const v8bf*)(p);
  v8bf hi = *(const v8bf*)(p + 8);
  return __builtin_shufflevector(lo, hi, 0,1,2,3,4,5,6,7,8,9,10,11,12,13,14,15);
}

// ---------------------------------------------------------------------------
// Kernel 1: fused QKV projection.  Out[m,n] = sum_k X[m,k]*W[n,k] + b[n]
//   z==0: fp32 q_a to out_qa AND bf16 (scaled 1/sqrt(dk)) to Qb
//   z==1: bf16 -> Kb      z==2: bf16 -> Vb
// bf16 tensors stored [B][H][S][dk] (head-contiguous).
// WG = 256 threads (8 waves). WG tile 128x128, wave tile 64x32, K-step 32.
// ---------------------------------------------------------------------------
__global__ void __launch_bounds__(256)
proj_kernel(const float* __restrict__ X,
            const float* __restrict__ Wq, const float* __restrict__ bq,
            const float* __restrict__ Wk, const float* __restrict__ bk,
            const float* __restrict__ Wv, const float* __restrict__ bv,
            float* __restrict__ qa_out,
            __bf16* __restrict__ Qb, __bf16* __restrict__ Kb,
            __bf16* __restrict__ Vb)
{
  const int z = blockIdx.z;
  const float* W    = (z == 0) ? Wq : (z == 1) ? Wk : Wv;
  const float* bias = (z == 0) ? bq : (z == 1) ? bk : bv;
  __bf16*      Ob   = (z == 0) ? Qb : (z == 1) ? Kb : Vb;

  const int nBase = blockIdx.x * 128;
  const int mBase = blockIdx.y * 128;
  const int tid   = threadIdx.x;
  const int lane  = tid & 31;
  const int wave  = tid >> 5;
  const int m00   = (wave >> 2) * 64;
  const int n00   = (wave & 3) * 32;

  __shared__ __bf16 lA[128][40];        // [m][k]
  __shared__ __bf16 lB[32][136];        // [k][n]  (transposed W tile)

  v8f acc[4][2];
  #pragma unroll
  for (int i = 0; i < 4; ++i)
    #pragma unroll
    for (int t = 0; t < 2; ++t) acc[i][t] = {};

  const int rowA  = tid >> 1;           // 0..127
  const int ksegA = (tid & 1) * 16;     // 0 / 16
  const int pB    = tid >> 2;           // 0..63  n-pair
  const int ksegB = (tid & 3) * 8;      // 0,8,16,24

  for (int kk = 0; kk < DDIM; kk += 32) {
    __syncthreads();
    // ---- stage A tile (fp32 -> bf16, vectorized) -----------------------
    {
      const float* src = X + (size_t)(mBase + rowA) * DDIM + kk + ksegA;
      #pragma unroll
      for (int j = 0; j < 16; j += 4) {
        v4f f = *(const v4f*)(src + j);
        lA[rowA][ksegA + j + 0] = (__bf16)f.x;
        lA[rowA][ksegA + j + 1] = (__bf16)f.y;
        lA[rowA][ksegA + j + 2] = (__bf16)f.z;
        lA[rowA][ksegA + j + 3] = (__bf16)f.w;
      }
    }
    // ---- stage W tile transposed, pair-packed b32 stores ---------------
    // thread owns n rows {2p, 2p+1}, packs (W[2p][k],W[2p+1][k]) -> b32
    {
      const float* s0 = W + (size_t)(nBase + 2 * pB)     * DDIM + kk + ksegB;
      const float* s1 = W + (size_t)(nBase + 2 * pB + 1) * DDIM + kk + ksegB;
      v4f f00 = *(const v4f*)(s0), f01 = *(const v4f*)(s0 + 4);
      v4f f10 = *(const v4f*)(s1), f11 = *(const v4f*)(s1 + 4);
      #pragma unroll
      for (int j = 0; j < 4; ++j) {
        union { __bf16 h[2]; unsigned u; } u0, u1;
        u0.h[0] = (__bf16)f00[j]; u0.h[1] = (__bf16)f10[j];
        u1.h[0] = (__bf16)f01[j]; u1.h[1] = (__bf16)f11[j];
        *(unsigned*)&lB[ksegB + j    ][2 * pB] = u0.u;
        *(unsigned*)&lB[ksegB + 4 + j][2 * pB] = u1.u;
      }
    }
    __syncthreads();

    if (kk + 32 < DDIM) {   // prefetch next K-slab -> global_prefetch_b8
      __builtin_prefetch(X + (size_t)(mBase + rowA) * DDIM + kk + 32 + ksegA, 0, 3);
      __builtin_prefetch(W + (size_t)(nBase + 2 * pB) * DDIM + kk + 32 + ksegB, 0, 3);
    }

    // ---- WMMA: 4 M-tiles x 2 N-tiles, K=32 -----------------------------
    v16bf a[4];
    #pragma unroll
    for (int i = 0; i < 4; ++i)
      a[i] = ld16bf(&lA[m00 + i * 16 + (lane & 15)][(lane >> 4) * 16]);
    #pragma unroll
    for (int t = 0; t < 2; ++t) {
      v16bf bf;   // n = lane&15, k = (lane>>4)*16 + s
      #pragma unroll
      for (int s = 0; s < 16; ++s) bf[s] = lB[(lane >> 4) * 16 + s][n00 + t * 16 + (lane & 15)];
      #pragma unroll
      for (int i = 0; i < 4; ++i)
        acc[i][t] = __builtin_amdgcn_wmma_f32_16x16x32_bf16(
            false, a[i], false, bf, (short)0, acc[i][t], false, false);
    }
  }

  // ---- epilogue: bias add, fp32 q_a + bf16 head-major stores -----------
  const int nLoc  = lane & 15;
  const int mHalf = lane >> 4;
  #pragma unroll
  for (int t = 0; t < 2; ++t) {
    const int   N  = nBase + n00 + t * 16 + nLoc;
    const float bN = bias[N];
    const int   h  = N >> 6, d = N & 63;
    #pragma unroll
    for (int i = 0; i < 4; ++i) {
      #pragma unroll
      for (int r = 0; r < 8; ++r) {
        const int M  = mBase + m00 + i * 16 + r + mHalf * 8;
        const int bb = M >> 10, s = M & 1023;
        const float val = acc[i][t][r] + bN;
        const size_t ob = ((size_t)(bb * NHEAD + h) * SEQ + s) * DK + d;
        if (z == 0) {
          qa_out[(size_t)M * DDIM + N] = val;
          Ob[ob] = (__bf16)(val * 0.125f);   // fold 1/sqrt(dk) into Q
        } else {
          Ob[ob] = (__bf16)val;
        }
      }
    }
  }
}

// ---------------------------------------------------------------------------
// Kernel 2: attention.  WG = 128 thr (4 waves), 64 q rows per WG, 16 per wave.
// Transposed-score formulation:  scores^T[ki][qi] = K · Q^T
//   - K and V staged row-major in LDS (pure b128 copies, no transpose)
//   - Q^T built once per WG; B-fragments live in registers all kernel
//   - e = (s>=0 || ki>=len) ? 0 : exp(s); lane holds consecutive-ki pairs
//     -> packed b32 stores straight into A-layout lE[qi][ki]
//   - ctx += e @ v ; rowsum is one scalar per lane (qi = lane&15)
// ---------------------------------------------------------------------------
__global__ void __launch_bounds__(128)
attn_kernel(const __bf16* __restrict__ Qb, const __bf16* __restrict__ Kb,
            const __bf16* __restrict__ Vb, const int* __restrict__ length,
            float* __restrict__ out_ctx)
{
  const int bh   = blockIdx.y;             // b*H + h
  const int b    = bh >> 4, h = bh & 15;
  const int tid  = threadIdx.x;
  const int lane = tid & 31;
  const int wave = tid >> 5;               // 0..3
  const int qBlk = blockIdx.x * 64;        // WG's q-row base
  const int qi0  = wave * 16;              // wave's local q base
  const int len  = length[b];

  const __bf16* qBase = Qb + (size_t)bh * (SEQ * DK);
  const __bf16* kBase = Kb + (size_t)bh * (SEQ * DK);
  const __bf16* vBase = Vb + (size_t)bh * (SEQ * DK);

  __shared__ __bf16 lQT[64][72];           // [d][qi]   (one-time transpose)
  __shared__ __bf16 lK [64][72];           // [ki][d]
  __shared__ __bf16 lV [64][72];           // [ki][d]
  __shared__ __bf16 lE [4][16][72];        // [wave][qi][ki]  A-layout e tile

  // ---- one-time: transpose 64x64 q tile into lQT (pair-packed b32) -----
  {
    const int p    = tid >> 2;             // 0..31 qi-pair
    const int dsg  = (tid & 3) * 16;       // 0,16,32,48
    const __bf16* q0 = qBase + (size_t)(qBlk + 2 * p)     * DK + dsg;
    const __bf16* q1 = qBase + (size_t)(qBlk + 2 * p + 1) * DK + dsg;
    v8bf a0 = *(const v8bf*)(q0), a1 = *(const v8bf*)(q0 + 8);
    v8bf b0 = *(const v8bf*)(q1), b1 = *(const v8bf*)(q1 + 8);
    #pragma unroll
    for (int j = 0; j < 8; ++j) {
      union { __bf16 h[2]; unsigned u; } u0, u1;
      u0.h[0] = a0[j]; u0.h[1] = b0[j];
      u1.h[0] = a1[j]; u1.h[1] = b1[j];
      *(unsigned*)&lQT[dsg + j    ][2 * p] = u0.u;
      *(unsigned*)&lQT[dsg + 8 + j][2 * p] = u1.u;
    }
  }
  __syncthreads();

  // B-fragments of Q^T: qi = lane&15, 16 d values; resident whole kernel
  v16bf bq[2];
  #pragma unroll
  for (int c = 0; c < 2; ++c)
    for (int s = 0; s < 16; ++s) bq[c][s] = lQT[c * 32 + (lane >> 4) * 16 + s][qi0 + (lane & 15)];

  v8f ctx[4];
  #pragma unroll
  for (int t = 0; t < 4; ++t) ctx[t] = {};
  float racc = 0.f;                        // rowsum partial for qi = lane&15

  const int kiLoc = tid >> 1;              // 0..63
  const int dseg  = (tid & 1) * 32;        // 0 / 32

  for (int kb = 0; kb < SEQ; kb += 64) {
    __syncthreads();
    // ---- stage K and V row-major: pure vector copies -------------------
    {
      const __bf16* ks = kBase + (size_t)(kb + kiLoc) * DK + dseg;
      const __bf16* vs = vBase + (size_t)(kb + kiLoc) * DK + dseg;
      #pragma unroll
      for (int j = 0; j < 32; j += 8) {
        *(v8bf*)&lK[kiLoc][dseg + j] = *(const v8bf*)(ks + j);
        *(v8bf*)&lV[kiLoc][dseg + j] = *(const v8bf*)(vs + j);
      }
    }
    __syncthreads();

    if (kb + 64 < SEQ) {
      __builtin_prefetch(kBase + (size_t)(kb + 64 + kiLoc) * DK + dseg, 0, 3);
      __builtin_prefetch(vBase + (size_t)(kb + 64 + kiLoc) * DK + dseg, 0, 3);
    }

    // ---- scores^T: 64 ki rows x 16 qi ----------------------------------
    v8f sc[4];
    #pragma unroll
    for (int t = 0; t < 4; ++t) sc[t] = {};
    #pragma unroll
    for (int c = 0; c < 2; ++c) {
      #pragma unroll
      for (int t = 0; t < 4; ++t) {
        v16bf ak = ld16bf(&lK[t * 16 + (lane & 15)][c * 32 + (lane >> 4) * 16]);
        sc[t] = __builtin_amdgcn_wmma_f32_16x16x32_bf16(
            false, ak, false, bq[c], (short)0, sc[t], false, false);
      }
    }

    // ---- mask/exp, pair-packed b32 stores into A-layout lE -------------
    // element (t,r,lane): ki = kb + t*16 + r + (lane>>4)*8, qi = lane&15
    #pragma unroll
    for (int t = 0; t < 4; ++t) {
      #pragma unroll
      for (int r = 0; r < 8; r += 2) {
        const int ki = kb + t * 16 + r + (lane >> 4) * 8;
        const float s0 = sc[t][r], s1 = sc[t][r + 1];
        const float e0 = (s0 < 0.f && ki     < len) ? __expf(s0) : 0.f;
        const float e1 = (s1 < 0.f && ki + 1 < len) ? __expf(s1) : 0.f;
        racc += e0 + e1;
        union { __bf16 h[2]; unsigned u; } pe;
        pe.h[0] = (__bf16)e0; pe.h[1] = (__bf16)e1;
        *(unsigned*)&lE[wave][lane & 15][t * 16 + (lane >> 4) * 8 + r] = pe.u;
      }
    }

    // ---- context accumulate: ctx += e @ v ------------------------------
    #pragma unroll
    for (int c = 0; c < 2; ++c) {
      v16bf ae = ld16bf(&lE[wave][lane & 15][c * 32 + (lane >> 4) * 16]);
      #pragma unroll
      for (int t = 0; t < 4; ++t) {
        v16bf bv;   // d = lane&15, ki = (lane>>4)*16 + s
        #pragma unroll
        for (int s = 0; s < 16; ++s) bv[s] = lV[c * 32 + (lane >> 4) * 16 + s][t * 16 + (lane & 15)];
        ctx[t] = __builtin_amdgcn_wmma_f32_16x16x32_bf16(
            false, ae, false, bv, (short)0, ctx[t], false, false);
      }
    }
  }

  // ---- rowsum: halves hold disjoint ki -> one butterfly step -----------
  racc += __shfl_xor(racc, 16, 32);
  const float rinv = 1.0f / (racc + 1e-8f);
  // redistribute: ctx rows need inv for qi = r + (lane>>4)*8
  float inv[8];
  #pragma unroll
  for (int r = 0; r < 8; ++r)
    inv[r] = __shfl(rinv, r + ((lane >> 4) << 3), 32);

  // ---- normalize + store context [B,S,H*dk] ----------------------------
  #pragma unroll
  for (int t = 0; t < 4; ++t) {
    const int dcol = h * DK + t * 16 + (lane & 15);
    #pragma unroll
    for (int r = 0; r < 8; ++r) {
      const int srow = qBlk + qi0 + r + (lane >> 4) * 8;
      out_ctx[(size_t)(b * SEQ + srow) * DDIM + dcol] = ctx[t][r] * inv[r];
    }
  }
}

// ---------------------------------------------------------------------------
extern "C" void kernel_launch(void* const* d_in, const int* in_sizes, int n_in,
                              void* d_out, int out_size, void* d_ws, size_t ws_size,
                              hipStream_t stream) {
  const float* Q      = (const float*)d_in[0];
  const int*   length = (const int*)d_in[1];
  const float* Wq     = (const float*)d_in[2];
  const float* bq     = (const float*)d_in[3];
  const float* Wk     = (const float*)d_in[4];
  const float* bk     = (const float*)d_in[5];
  const float* Wv     = (const float*)d_in[6];
  const float* bv     = (const float*)d_in[7];

  float* out_ctx = (float*)d_out;                        // [4,1024,1024]
  float* out_qa  = out_ctx + (size_t)4 * SEQ * DDIM;     // [4,1024,1024]

  // workspace: 3 bf16 tensors [B,H,S,dk] = 3 * 8 MB = 24 MB
  __bf16* Qb = (__bf16*)d_ws;
  __bf16* Kb = Qb + (size_t)4 * NHEAD * SEQ * DK;
  __bf16* Vb = Kb + (size_t)4 * NHEAD * SEQ * DK;

  dim3 pgrid(DDIM / 128, (4 * SEQ) / 128, 3);            // 8 x 32 x 3
  proj_kernel<<<pgrid, dim3(256), 0, stream>>>(Q, Wq, bq, Wk, bk, Wv, bv,
                                               out_qa, Qb, Kb, Vb);

  dim3 agrid(SEQ / 64, 4 * NHEAD);                       // 16 x 64
  attn_kernel<<<agrid, dim3(128), 0, stream>>>(Qb, Kb, Vb, length, out_ctx);
}
